// OperationLibrary_5111011082884
// MI455X (gfx1250) — matrix operand order repaired and checked by reading, not compiled, not run on hardware
//
#include <hip/hip_runtime.h>
#include <hip/hip_bf16.h>
#include <stdint.h>

#define N_OPS 32
#define D 128
#define TOKENS 16384      // 8 * 2048
#define PAIRS 32768       // TOKENS * 2
#define MAX_TILES 2080    // 2048 data tiles + up to 32 padding tiles

typedef __bf16 bf16_t;
typedef bf16_t v16bf __attribute__((ext_vector_type(16)));
typedef bf16_t v8bf  __attribute__((ext_vector_type(8)));
typedef float  v8f   __attribute__((ext_vector_type(8)));

union Frag {
    v16bf v;
    v8bf  h[2];
    unsigned short u[16];
};

__device__ inline unsigned short f32_to_bf16_rne(float f) {
    unsigned int u = __float_as_uint(f);
    u += 0x7FFFu + ((u >> 16) & 1u);
    return (unsigned short)(u >> 16);
}

// Fragment layouts used here (wave32):
//  A (16x32 MxK): lane<16 -> row M=lane, K {kh*16..kh*16+7} in elems 0..7 and
//                 K {kh*16+8..} in elems 8..15 (kh = lane>>4). Row-major bf16
//                 memory is therefore fragment-ready: two 16B spans per K-step.
//  B (32x16 KxN): lanes 0-15 hold K 0..15 (2/VGPR), lanes 16-31 hold K 16..31 (this code's order).
//                 tq/wleak are pre-swizzled so a fragment is one 32B span/lane:
//                 frag f = ((nt*4 + s)*32 + lane)*16 + e, with
//                 d = 32*s + (lane>>4)*16 + e,  o = nt*16 + (lane&15).

// ---------------------------------------------------------------- init
__global__ void init_kernel(int* hist, int* cursor, int* list) {
    int id = blockIdx.x * 256 + threadIdx.x;
    if (id < N_OPS) { hist[id] = 0; cursor[id] = 0; }
    if (id < MAX_TILES * 16) list[id] = -1;
}

// ------------------------------------------------ split x into bf16 hi + lo
// hi = truncate(f) (exact remainder captured by lo, so RNE not needed);
// hi pair packed with one v_perm_b32.
__global__ __launch_bounds__(256) void prep_x(const float* __restrict__ x,
                                              unsigned int* __restrict__ xhi,
                                              unsigned int* __restrict__ xlo) {
    int i = blockIdx.x * 256 + threadIdx.x;       // one thread = 2 elements
    if (i >= TOKENS * D / 2) return;
    float f0 = x[2 * i], f1 = x[2 * i + 1];
    unsigned int b0 = __float_as_uint(f0), b1 = __float_as_uint(f1);
    xhi[i] = __builtin_amdgcn_perm(b1, b0, 0x07060302u);  // {b1.hi16, b0.hi16}
    float h0 = __uint_as_float(b0 & 0xFFFF0000u);
    float h1 = __uint_as_float(b1 & 0xFFFF0000u);
    unsigned int l0 = f32_to_bf16_rne(f0 - h0);
    unsigned int l1 = f32_to_bf16_rne(f1 - h1);
    xlo[i] = l0 | (l1 << 16);
}

// ------------------------------------------------ per-expert ternary quantize
// Emits tq in B-fragment-swizzled order (see map above).
__global__ __launch_bounds__(256) void quantize_kernel(const float* __restrict__ ops,
                                                       unsigned short* __restrict__ tq,
                                                       float* __restrict__ scales) {
    __shared__ float red[256];
    __shared__ float sscale;
    int n = blockIdx.x, tid = threadIdx.x;
    const float* w = ops + (size_t)n * D * D;
    float acc = 0.f;
    for (int e = tid; e < D * D; e += 256) acc += fabsf(w[e]);
    red[tid] = acc; __syncthreads();
    for (int s = 128; s > 0; s >>= 1) { if (tid < s) red[tid] += red[tid + s]; __syncthreads(); }
    if (tid == 0) {
        float sc = fmaxf(red[0] * (1.0f / (D * D)), 1e-5f);
        sscale = sc; scales[n] = sc;
    }
    __syncthreads();
    float inv = 1.0f / sscale;
    unsigned short* tqn = tq + (size_t)n * D * D;
    for (int f = tid; f < D * D; f += 256) {
        int e    = f & 15;
        int lane = (f >> 4) & 31;
        int s    = (f >> 9) & 3;
        int nt   = f >> 11;
        int d = 32 * s + (lane >> 4) * 16 + e;
        int o = nt * 16 + (lane & 15);
        float q = rintf(w[(size_t)o * D + d] * inv);    // round-half-even like jnp.round
        q = fminf(1.f, fmaxf(-1.f, q));
        tqn[f] = (q == 0.f) ? 0u : (q > 0.f ? 0x3F80u : 0xBF80u);
    }
}

// -------------------- W_leak = (LEAK/N) * sum_n scale_n * t_n  (same frag layout)
// The frag-index -> (d,o) map is shared with tq, so this is an index-identity sum.
__global__ __launch_bounds__(256) void wleak_kernel(const unsigned short* __restrict__ tq,
                                                    const float* __restrict__ scales,
                                                    unsigned short* __restrict__ wleak) {
    __shared__ float ss[N_OPS];
    if (threadIdx.x < N_OPS) ss[threadIdx.x] = scales[threadIdx.x];
    __syncthreads();
    int id = blockIdx.x * 256 + threadIdx.x;
    if (id >= D * D) return;
    float acc = 0.f;
    for (int n = 0; n < N_OPS; ++n) {
        unsigned short us = tq[(size_t)n * D * D + id];
        if (us) acc += (us & 0x8000u) ? -ss[n] : ss[n];
    }
    wleak[id] = f32_to_bf16_rne(acc * (1e-5f / (float)N_OPS));
}

// ---------------------------------------------------------------- routing build
__global__ void hist_kernel(const long long* __restrict__ idx, int* hist) {
    int i = blockIdx.x * 256 + threadIdx.x;
    if (i < PAIRS) atomicAdd(&hist[(int)idx[i]], 1);
}

__global__ void scan_kernel(const int* __restrict__ hist, int* tileBase) {
    if (threadIdx.x == 0 && blockIdx.x == 0) {
        int tb = 0;
        for (int n = 0; n < N_OPS; ++n) { tileBase[n] = tb; tb += (hist[n] + 15) >> 4; }
        tileBase[N_OPS] = tb;
    }
}

__global__ void fill_kernel(const long long* __restrict__ idx,
                            const int* __restrict__ tileBase, int* cursor, int* list) {
    int i = blockIdx.x * 256 + threadIdx.x;
    if (i < PAIRS) {
        int n = (int)idx[i];
        int pos = atomicAdd(&cursor[n], 1);
        list[(size_t)tileBase[n] * 16 + pos] = i;   // pair index; token = i>>1
    }
}

// ------------------------------------------------------------- fragment loads
__device__ inline void load_a(const unsigned short* __restrict__ xrow,
                              int s, int kh, Frag& a) {
    const bf16_t* p = (const bf16_t*)xrow + 32 * s + kh * 16;
    a.h[0] = *(const v8bf*)(p);        // K = 32s + kh*16 .. +7
    a.h[1] = *(const v8bf*)(p + 8);    // K = 32s + kh*16 + 8 .. +15
}

__device__ inline void load_b(const unsigned short* __restrict__ frags,
                              int nt, int s, int lane, Frag& b) {
    b.v = *(const v16bf*)((const bf16_t*)frags + (((nt * 4 + s) * 32 + lane) << 4));
}

// ------------------------------------------------------------- leak GEMM
// out[tile](16x128) = x[tile] @ W_leak.T, initializes out.
__global__ __launch_bounds__(32) void leak_gemm(const unsigned short* __restrict__ xhi,
                                                const unsigned short* __restrict__ wleak,
                                                float* __restrict__ out) {
    int tile = blockIdx.x;
    int lane = threadIdx.x;
    int m = lane & 15, kh = lane >> 4;
    const unsigned short* xrow = xhi + ((size_t)tile * 16 + m) * D;
    Frag a[4];
#pragma unroll
    for (int s = 0; s < 4; ++s) load_a(xrow, s, kh, a[s]);
#pragma unroll
    for (int nt = 0; nt < 8; ++nt) {
        v8f c = {};
#pragma unroll
        for (int s = 0; s < 4; ++s) {
            Frag b;
            load_b(wleak, nt, s, lane, b);
            c = __builtin_amdgcn_wmma_f32_16x16x32_bf16(false, a[s].v, false, b.v,
                                                        (short)0, c, false, false);
        }
        float* ocol = out + (size_t)tile * 16 * D + nt * 16 + m;
#pragma unroll
        for (int r = 0; r < 8; ++r) {
            int M = r + 8 * kh;
            ocol[(size_t)M * D] = c[r];
        }
    }
}

// ------------------------------------------------------------- routed grouped GEMM
// One wave per 16 (token,expert) pairs of one expert; hi+lo bf16 chained into the
// same f32 accumulator -> fp32-accurate; row-scaled atomic scatter.
__global__ __launch_bounds__(32) void routed_gemm(const unsigned short* __restrict__ xhi,
                                                  const unsigned short* __restrict__ xlo,
                                                  const unsigned short* __restrict__ tq,
                                                  const float* __restrict__ scales,
                                                  const float* __restrict__ wpair,
                                                  const int* __restrict__ tileBase,
                                                  const int* __restrict__ list,
                                                  float* out) {
    __shared__ int   sTB[N_OPS + 1];
    __shared__ int   stok[16];
    __shared__ float scoef[16];
    int lane = threadIdx.x;
    for (int i = lane; i < N_OPS + 1; i += 32) sTB[i] = tileBase[i];
    __syncthreads();
    int bid = blockIdx.x;
    if (bid >= sTB[N_OPS]) return;            // wave-uniform exit (EXEC stays all-1)
    int n = 0;
    while (n < N_OPS - 1 && bid >= sTB[n + 1]) ++n;
    float scale_n = scales[n];
    if (lane < 16) {
        int e = list[(size_t)bid * 16 + lane];
        if (e >= 0) { stok[lane] = e >> 1; scoef[lane] = wpair[e] * scale_n; }
        else        { stok[lane] = -1;     scoef[lane] = 0.f; }
    }
    __syncthreads();

    int m = lane & 15, kh = lane >> 4;
    int tokm = stok[m];
    size_t row = (size_t)(tokm < 0 ? 0 : tokm) * D;
    Frag ahi[4], alo[4];
#pragma unroll
    for (int s = 0; s < 4; ++s) {
        load_a(xhi + row, s, kh, ahi[s]);
        load_a(xlo + row, s, kh, alo[s]);
    }
    const unsigned short* tqn = tq + (size_t)n * D * D;
#pragma unroll
    for (int nt = 0; nt < 8; ++nt) {
        v8f c = {};
#pragma unroll
        for (int s = 0; s < 4; ++s) {
            Frag b;
            load_b(tqn, nt, s, lane, b);
            c = __builtin_amdgcn_wmma_f32_16x16x32_bf16(false, ahi[s].v, false, b.v,
                                                        (short)0, c, false, false);
            c = __builtin_amdgcn_wmma_f32_16x16x32_bf16(false, alo[s].v, false, b.v,
                                                        (short)0, c, false, false);
        }
#pragma unroll
        for (int r = 0; r < 8; ++r) {
            int M = r + 8 * kh;
            int tk = stok[M];
            if (tk >= 0)
                atomicAdd(&out[(size_t)tk * D + nt * 16 + m], c[r] * scoef[M]);
        }
    }
}

// ---------------------------------------------------------------- launch
extern "C" void kernel_launch(void* const* d_in, const int* in_sizes, int n_in,
                              void* d_out, int out_size, void* d_ws, size_t ws_size,
                              hipStream_t stream) {
    const float*     x       = (const float*)d_in[0];
    const long long* indices = (const long long*)d_in[1];   // int64 in reference
    const float*     weights = (const float*)d_in[2];
    const float*     ops     = (const float*)d_in[3];
    float*           out     = (float*)d_out;

    char* ws = (char*)d_ws;
    unsigned short* tq       = (unsigned short*)(ws);            // 1,048,576 B (frag layout)
    unsigned short* wleak    = (unsigned short*)(ws + 1048576);  // 32,768 B  (frag layout)
    float*          scales   = (float*)(ws + 1081344);           // 128 B
    int*            hist     = (int*)(ws + 1081472);             // 128 B
    int*            cursor   = (int*)(ws + 1081600);             // 128 B
    int*            tileBase = (int*)(ws + 1081728);             // 256 B
    int*            list     = (int*)(ws + 1081984);             // 133,120 B
    unsigned short* xhi      = (unsigned short*)(ws + 1215232);  // 4,194,304 B
    unsigned short* xlo      = (unsigned short*)(ws + 5409536);  // 4,194,304 B

    init_kernel<<<(MAX_TILES * 16 + 255) / 256, 256, 0, stream>>>(hist, cursor, list);
    prep_x<<<(TOKENS * D / 2 + 255) / 256, 256, 0, stream>>>(x, (unsigned int*)xhi,
                                                             (unsigned int*)xlo);
    quantize_kernel<<<N_OPS, 256, 0, stream>>>(ops, tq, scales);
    wleak_kernel<<<(D * D + 255) / 256, 256, 0, stream>>>(tq, scales, wleak);
    hist_kernel<<<(PAIRS + 255) / 256, 256, 0, stream>>>(indices, hist);
    scan_kernel<<<1, 32, 0, stream>>>(hist, tileBase);
    fill_kernel<<<(PAIRS + 255) / 256, 256, 0, stream>>>(indices, tileBase, cursor, list);
    leak_gemm<<<TOKENS / 16, 32, 0, stream>>>(xhi, wleak, out);
    routed_gemm<<<MAX_TILES, 32, 0, stream>>>(xhi, xlo, tq, scales, weights,
                                              tileBase, list, out);
}
